// ChemGeomFeatEncoder_51866025066849
// MI455X (gfx1250) — matrix operand order repaired and checked by reading, not compiled, not run on hardware
//
#include <hip/hip_runtime.h>
#include <cstdint>
#include <cstddef>

// ---------------------------------------------------------------------------
// CDNA5 / gfx1250 fused MLP encoder.
// wave32, WMMA 16x16x32 f16 (fp32 accumulate), 64 rows per block,
// async global->LDS weight staging when the toolchain exposes it.
// ---------------------------------------------------------------------------

typedef __attribute__((ext_vector_type(16))) _Float16 v16h;
typedef __attribute__((ext_vector_type(8)))  float    v8f;
typedef __attribute__((ext_vector_type(4)))  int      i32x4;

#define HDIM   128
#define NVERTS 100000
#define NGRAPH 20000
#define NNBR   500000

#if defined(__HIP_DEVICE_COMPILE__) && \
    __has_builtin(__builtin_amdgcn_global_load_async_to_lds_b128)
#define USE_ASYNC_LDS 1
typedef __attribute__((address_space(1))) i32x4 i32x4_as1;   // global
typedef __attribute__((address_space(3))) i32x4 i32x4_as3;   // LDS
#else
#define USE_ASYNC_LDS 0
#endif

// A-operand (16x32 f16) per-lane K permutation within a 32-K group.
// Lane l (half index hh in 0..15): K = hh + 16*(l>>4).
// Inverse: storage position of K=kk is kk (identity).
__device__ __forceinline__ int kpermA(int kk) {
    return kk;
}
// B-operand (32x16 f16): lane l holds column n=l%16, K = hh + 16*(l>>4)
// within the group -> identity permutation (store K-groups in natural order).

// Fast activations: single v_rcp_f32 / v_exp_f32 / v_log_f32, no IEEE div path.
__device__ __forceinline__ float fsigmoid(float x) {
    return __builtin_amdgcn_rcpf(1.0f + __expf(-x));
}
__device__ __forceinline__ float fsilu(float x) { return x * fsigmoid(x); }
__device__ __forceinline__ float fsoftplus(float x) {
    return (x > 15.0f) ? x : __logf(1.0f + __expf(x));
}

// Copy 16B-granular packed data into LDS. Async path writes LDS directly
// (ASYNCcnt), overlapping with subsequent VGPR work; caller must
// async_join() + __syncthreads() before consuming.
__device__ __forceinline__ void stage_to_lds(_Float16* dstLds, const _Float16* src,
                                             int nHalves, int tid) {
#if USE_ASYNC_LDS
    for (int i = tid; i < nHalves / 8; i += 256) {
        __builtin_amdgcn_global_load_async_to_lds_b128(
            (i32x4_as1*)(const void*)(src + i * 8),
            (i32x4_as3*)(void*)(dstLds + i * 8),
            0, 0);
    }
#else
    const uint4* s = (const uint4*)src;
    uint4* t = (uint4*)dstLds;
    for (int i = tid; i < nHalves / 8; i += 256) t[i] = s[i];
#endif
}

__device__ __forceinline__ void async_join() {
#if USE_ASYNC_LDS
#if __has_builtin(__builtin_amdgcn_s_wait_asynccnt)
    __builtin_amdgcn_s_wait_asynccnt(0);
#else
    asm volatile("s_wait_asynccnt 0" ::: "memory");
#endif
#endif
}

// ---------------------------------------------------------------------------
__global__ void zero_f32(float* __restrict__ p, long long n) {
    long long i = (long long)blockIdx.x * blockDim.x + threadIdx.x;
    if (i < n) p[i] = 0.0f;
}

// Pack fp32 weight W[K][N] (row-major, as in x @ W) into f16 WMMA-B layout,
// zero-padding K up to KPAD (multiple of 32):
//   dst[((k>>5)*N + n)*32 + (k&31)] = (f16)W[k*N + n]
__global__ void pack_weight(const float* __restrict__ W, int K, int N, int KPAD,
                            _Float16* __restrict__ dst) {
    int i = blockIdx.x * blockDim.x + threadIdx.x;
    if (i >= KPAD * N) return;
    int k = i / N, n = i % N;
    float val = (k < K) ? W[k * N + n] : 0.0f;
    dst[(((k >> 5) * N + n) << 5) + (k & 31)] = (_Float16)val;
}

// Fold Linear bias + eval-mode BatchNorm into y = gemm*scale + off.
//   scale = g * rsqrt(v + eps);  off = (b_lin - m)*scale + b_bn
__global__ void prep_affine(const float* __restrict__ blin, const float* __restrict__ g,
                            const float* __restrict__ bbn,  const float* __restrict__ m,
                            const float* __restrict__ var,  int N,
                            float* __restrict__ scale, float* __restrict__ off) {
    for (int i = threadIdx.x; i < N; i += blockDim.x) {
        float s = g[i] * rsqrtf(var[i] + 1e-5f);
        scale[i] = s;
        off[i]   = (blin[i] - m[i]) * s + bbn[i];
    }
}

// ---------------------------------------------------------------------------
// Fused MLP: X[ROWS tile] -> GEMM1(K=D_IN_PAD) -> BN+SiLU -> GEMM2(K=128)
//            -> BN -> (MODE 0: store rows; MODE 1: gate + atomic segment-sum)
// Block = 256 threads = 8 waves; wave w owns output columns [16w, 16w+16) and
// iterates over MT row-tiles (ROWS = 16*MT), reusing the B fragment across
// all MT v_wmma issues per K-step.
// Input may be a virtual concat of X (dx cols) and X2 (dx2 cols).
// ---------------------------------------------------------------------------
template <int D_IN_PAD, int D_OUT, int MODE, int MT>
__global__ __launch_bounds__(256) void mlp_fused(
    const float* __restrict__ X,  int ldx,  int dx,
    const float* __restrict__ X2, int ldx2, int dx2,
    const _Float16* __restrict__ Wp1,
    const float* __restrict__ sc1, const float* __restrict__ of1,
    const _Float16* __restrict__ Wp2,
    const float* __restrict__ sc2, const float* __restrict__ of2,
    float* __restrict__ out, const int* __restrict__ vids, int nrows) {

    constexpr int ROWS = 16 * MT;

    __shared__ __align__(32) _Float16 sW1[D_IN_PAD * HDIM];
    __shared__ __align__(32) _Float16 sW2[HDIM * D_OUT];
    __shared__ __align__(32) _Float16 sA[ROWS * D_IN_PAD];
    __shared__ __align__(32) _Float16 sH[ROWS * HDIM];

    const int tid = threadIdx.x;
    const int w  = tid >> 5;   // wave id 0..7
    const int l  = tid & 31;   // lane
    const int lh = l >> 4;     // lane half
    const int lm = l & 15;

    // Kick off weight staging (async: LDS written directly, ASYNCcnt-tracked,
    // overlapped with the fp32->f16 A-tile conversion below).
    stage_to_lds(sW1, Wp1, D_IN_PAD * HDIM, tid);
    stage_to_lds(sW2, Wp2, HDIM * D_OUT, tid);

    const long long row0 = (long long)blockIdx.x * ROWS;

    // Load ROWS x D_IN_PAD input tile, fp32 -> f16, into WMMA-A layout.
    for (int idx = tid; idx < ROWS * D_IN_PAD; idx += 256) {
        int m = idx / D_IN_PAD;
        int k = idx % D_IN_PAD;
        long long r = row0 + m;
        float val = 0.0f;
        if (r < nrows) {
            if (k < dx)            val = X[r * (long long)ldx + k];
            else if (k < dx + dx2) val = X2[r * (long long)ldx2 + (k - dx)];
        }
        sA[(((k >> 5) * ROWS + m) << 5) + kpermA(k & 31)] = (_Float16)val;
    }
    async_join();
    __syncthreads();

    // GEMM1: A[ROWS x D_IN_PAD] x W1[D_IN_PAD x 128]
    v8f c[MT];
#pragma unroll
    for (int mt = 0; mt < MT; ++mt) c[mt] = {};
#pragma unroll
    for (int kb = 0; kb < D_IN_PAD / 32; ++kb) {
        v16h b = *(const v16h*)&sW1[((kb * HDIM + (w * 16 + lm)) << 5) + lh * 16];
#pragma unroll
        for (int mt = 0; mt < MT; ++mt) {
            v16h a = *(const v16h*)&sA[((kb * ROWS + mt * 16 + lm) << 5) + lh * 16];
            c[mt] = __builtin_amdgcn_wmma_f32_16x16x32_f16(false, a, false, b,
                                                           (short)0, c[mt], false, false);
        }
    }

    // BN + SiLU; deposit hidden tiles in WMMA-A layout (hidden idx n == K of GEMM2).
    {
        const int n = w * 16 + lm;
        const float s1 = sc1[n], o1 = of1[n];
        const int gbase = ((n >> 5) * ROWS) << 5;
        const int kp = kpermA(n & 31);
#pragma unroll
        for (int mt = 0; mt < MT; ++mt) {
#pragma unroll
            for (int e = 0; e < 8; ++e) {
                int m = mt * 16 + e + lh * 8;
                float x = c[mt][e] * s1 + o1;
                sH[gbase + (m << 5) + kp] = (_Float16)fsilu(x);
            }
        }
    }
    __syncthreads();

    if constexpr (MODE == 0) {
        v8f d[MT];
#pragma unroll
        for (int mt = 0; mt < MT; ++mt) d[mt] = {};
#pragma unroll
        for (int kb = 0; kb < HDIM / 32; ++kb) {
            v16h b = *(const v16h*)&sW2[((kb * D_OUT + (w * 16 + lm)) << 5) + lh * 16];
#pragma unroll
            for (int mt = 0; mt < MT; ++mt) {
                v16h a = *(const v16h*)&sH[((kb * ROWS + mt * 16 + lm) << 5) + lh * 16];
                d[mt] = __builtin_amdgcn_wmma_f32_16x16x32_f16(false, a, false, b,
                                                               (short)0, d[mt], false, false);
            }
        }
        const int n = w * 16 + lm;
        const float s2 = sc2[n], o2 = of2[n];
#pragma unroll
        for (int mt = 0; mt < MT; ++mt) {
#pragma unroll
            for (int e = 0; e < 8; ++e) {
                long long r = row0 + mt * 16 + e + lh * 8;
                if (r < nrows) out[r * (long long)D_OUT + n] = d[mt][e] * s2 + o2;
            }
        }
    } else {
        // D_OUT == 256. Wave w: filter cols [16w,16w+16), core cols 128 + same.
        v8f df[MT], dc[MT];
#pragma unroll
        for (int mt = 0; mt < MT; ++mt) { df[mt] = {}; dc[mt] = {}; }
        const int nf = w * 16 + lm;
        const int nc = HDIM + w * 16 + lm;
#pragma unroll
        for (int kb = 0; kb < HDIM / 32; ++kb) {
            v16h bf = *(const v16h*)&sW2[((kb * D_OUT + nf) << 5) + lh * 16];
            v16h bc = *(const v16h*)&sW2[((kb * D_OUT + nc) << 5) + lh * 16];
#pragma unroll
            for (int mt = 0; mt < MT; ++mt) {
                v16h a = *(const v16h*)&sH[((kb * ROWS + mt * 16 + lm) << 5) + lh * 16];
                df[mt] = __builtin_amdgcn_wmma_f32_16x16x32_f16(false, a, false, bf,
                                                                (short)0, df[mt], false, false);
                dc[mt] = __builtin_amdgcn_wmma_f32_16x16x32_f16(false, a, false, bc,
                                                                (short)0, dc[mt], false, false);
            }
        }
        const float sF = sc2[nf], oF = of2[nf];
        const float sC = sc2[nc], oC = of2[nc];
#pragma unroll
        for (int mt = 0; mt < MT; ++mt) {
#pragma unroll
            for (int e = 0; e < 8; ++e) {
                long long r = row0 + mt * 16 + e + lh * 8;
                if (r < nrows) {
                    float yf = df[mt][e] * sF + oF;
                    float yc = dc[mt][e] * sC + oC;
                    float g  = fsigmoid(yf) * fsoftplus(yc);
                    atomicAdd(&out[(long long)vids[r] * HDIM + (w * 16 + lm)], g);
                }
            }
        }
    }
}

// ---------------------------------------------------------------------------
// Host side
// ---------------------------------------------------------------------------
struct MlpP {
    const float *W1, *W2, *b1, *b2;
    const float *bn1b, *bn1g, *bn1m, *bn1v;
    const float *bn2b, *bn2g, *bn2m, *bn2v;
};

// params tree is flattened in sorted-key order per MLP:
// W1, W2, b1, b2, bn1{b,g,m,v}, bn2{b,g,m,v}  (12 leaves)
static MlpP get_mlp(void* const* d_in, int base) {
    MlpP p;
    p.W1   = (const float*)d_in[base + 0];
    p.W2   = (const float*)d_in[base + 1];
    p.b1   = (const float*)d_in[base + 2];
    p.b2   = (const float*)d_in[base + 3];
    p.bn1b = (const float*)d_in[base + 4];
    p.bn1g = (const float*)d_in[base + 5];
    p.bn1m = (const float*)d_in[base + 6];
    p.bn1v = (const float*)d_in[base + 7];
    p.bn2b = (const float*)d_in[base + 8];
    p.bn2g = (const float*)d_in[base + 9];
    p.bn2m = (const float*)d_in[base + 10];
    p.bn2v = (const float*)d_in[base + 11];
    return p;
}

extern "C" void kernel_launch(void* const* d_in, const int* in_sizes, int n_in,
                              void* d_out, int out_size, void* d_ws, size_t ws_size,
                              hipStream_t stream) {
    (void)in_sizes; (void)n_in; (void)out_size; (void)ws_size;

    const float* geom_x = (const float*)d_in[0];   // [V, 22]
    const float* chem_x = (const float*)d_in[1];   // [G, 37]
    const float* surf_x = (const float*)d_in[2];   // [E, 60]
    const int*   vids   = (const int*)d_in[3];     // [E]

    // MLP param dicts, top-level sorted: chem_mlp, feat_mlp, geom_mlp, surf_chem_mlp
    MlpP chem = get_mlp(d_in, 4);
    MlpP feat = get_mlp(d_in, 16);
    MlpP geom = get_mlp(d_in, 28);
    MlpP surf = get_mlp(d_in, 40);

    float* out_hgeom = (float*)d_out;                              // [V,128]
    float* out_hchem = (float*)d_out + (size_t)NVERTS * HDIM;      // [G,128]

    // Workspace carving (~103 MB)
    char* wsb = (char*)d_ws;
    size_t off = 0;
    auto carve = [&](size_t bytes) { void* p = wsb + off; off += (bytes + 15) & ~(size_t)15; return p; };
    const size_t VH = (size_t)NVERTS * HDIM;
    float* agg  = (float*)carve(VH * 4);
    float* hmid = (float*)carve(VH * 4);
    _Float16* chemW1p = (_Float16*)carve((size_t)64  * 128 * 2);
    _Float16* chemW2p = (_Float16*)carve((size_t)128 * 128 * 2);
    _Float16* geomW1p = (_Float16*)carve((size_t)32  * 128 * 2);
    _Float16* geomW2p = (_Float16*)carve((size_t)128 * 128 * 2);
    _Float16* surfW1p = (_Float16*)carve((size_t)64  * 128 * 2);
    _Float16* surfW2p = (_Float16*)carve((size_t)128 * 256 * 2);
    _Float16* featW1p = (_Float16*)carve((size_t)256 * 128 * 2);
    _Float16* featW2p = (_Float16*)carve((size_t)128 * 128 * 2);
    float* chemS1 = (float*)carve(128*4); float* chemO1 = (float*)carve(128*4);
    float* chemS2 = (float*)carve(128*4); float* chemO2 = (float*)carve(128*4);
    float* geomS1 = (float*)carve(128*4); float* geomO1 = (float*)carve(128*4);
    float* geomS2 = (float*)carve(128*4); float* geomO2 = (float*)carve(128*4);
    float* surfS1 = (float*)carve(128*4); float* surfO1 = (float*)carve(128*4);
    float* surfS2 = (float*)carve(256*4); float* surfO2 = (float*)carve(256*4);
    float* featS1 = (float*)carve(128*4); float* featO1 = (float*)carve(128*4);
    float* featS2 = (float*)carve(128*4); float* featO2 = (float*)carve(128*4);

    auto cdiv = [](long long a, long long b) { return (int)((a + b - 1) / b); };

    // 1) zero the segment-sum accumulator
    zero_f32<<<cdiv(VH, 256), 256, 0, stream>>>(agg, (long long)VH);

    // 2) pack weights to f16 WMMA-B layout + fold bias/BN affines
    pack_weight<<<cdiv(64*128, 256),  256, 0, stream>>>(chem.W1,  37, 128,  64, chemW1p);
    pack_weight<<<cdiv(128*128, 256), 256, 0, stream>>>(chem.W2, 128, 128, 128, chemW2p);
    pack_weight<<<cdiv(32*128, 256),  256, 0, stream>>>(geom.W1,  22, 128,  32, geomW1p);
    pack_weight<<<cdiv(128*128, 256), 256, 0, stream>>>(geom.W2, 128, 128, 128, geomW2p);
    pack_weight<<<cdiv(64*128, 256),  256, 0, stream>>>(surf.W1,  60, 128,  64, surfW1p);
    pack_weight<<<cdiv(128*256, 256), 256, 0, stream>>>(surf.W2, 128, 256, 128, surfW2p);
    pack_weight<<<cdiv(256*128, 256), 256, 0, stream>>>(feat.W1, 256, 128, 256, featW1p);
    pack_weight<<<cdiv(128*128, 256), 256, 0, stream>>>(feat.W2, 128, 128, 128, featW2p);

    prep_affine<<<1, 256, 0, stream>>>(chem.b1, chem.bn1g, chem.bn1b, chem.bn1m, chem.bn1v, 128, chemS1, chemO1);
    prep_affine<<<1, 256, 0, stream>>>(chem.b2, chem.bn2g, chem.bn2b, chem.bn2m, chem.bn2v, 128, chemS2, chemO2);
    prep_affine<<<1, 256, 0, stream>>>(geom.b1, geom.bn1g, geom.bn1b, geom.bn1m, geom.bn1v, 128, geomS1, geomO1);
    prep_affine<<<1, 256, 0, stream>>>(geom.b2, geom.bn2g, geom.bn2b, geom.bn2m, geom.bn2v, 128, geomS2, geomO2);
    prep_affine<<<1, 256, 0, stream>>>(surf.b1, surf.bn1g, surf.bn1b, surf.bn1m, surf.bn1v, 128, surfS1, surfO1);
    prep_affine<<<1, 256, 0, stream>>>(surf.b2, surf.bn2g, surf.bn2b, surf.bn2m, surf.bn2v, 256, surfS2, surfO2);
    prep_affine<<<1, 256, 0, stream>>>(feat.b1, feat.bn1g, feat.bn1b, feat.bn1m, feat.bn1v, 128, featS1, featO1);
    prep_affine<<<1, 256, 0, stream>>>(feat.b2, feat.bn2g, feat.bn2b, feat.bn2m, feat.bn2v, 128, featS2, featO2);

    // 3) geom MLP: [V,22] -> hmid [V,128]
    mlp_fused<32, 128, 0, 4><<<cdiv(NVERTS, 64), 256, 0, stream>>>(
        geom_x, 22, 22, nullptr, 0, 0,
        geomW1p, geomS1, geomO1, geomW2p, geomS2, geomO2,
        hmid, nullptr, NVERTS);

    // 4) chem MLP: [G,37] -> d_out second tensor [G,128]
    mlp_fused<64, 128, 0, 4><<<cdiv(NGRAPH, 64), 256, 0, stream>>>(
        chem_x, 37, 37, nullptr, 0, 0,
        chemW1p, chemS1, chemO1, chemW2p, chemS2, chemO2,
        out_hchem, nullptr, NGRAPH);

    // 5) surf MLP + gate + atomic segment-sum: [E,60] -> agg [V,128]
    mlp_fused<64, 256, 1, 4><<<cdiv(NNBR, 64), 256, 0, stream>>>(
        surf_x, 60, 60, nullptr, 0, 0,
        surfW1p, surfS1, surfO1, surfW2p, surfS2, surfO2,
        agg, vids, NNBR);

    // 6) feat MLP on virtual concat(agg, hmid): [V,256] -> d_out first tensor
    mlp_fused<256, 128, 0, 4><<<cdiv(NVERTS, 64), 256, 0, stream>>>(
        agg, 128, 128, hmid, 128, 128,
        featW1p, featS1, featO1, featW2p, featS2, featO2,
        out_hgeom, nullptr, NVERTS);
}
